// AttentionTSP_2035814498412
// MI455X (gfx1250) — matrix operand order repaired and checked by reading, not compiled, not run on hardware
//
#include <hip/hip_runtime.h>
#include <hip/hip_bf16.h>
#include <math.h>

#define Bsz 256
#define Ssz 128
#define Esz 256
#define HIDsz 512
#define Hh 8
#define DKsz 32
#define NEGV -1e9f

typedef __attribute__((ext_vector_type(16))) _Float16 v16h;
typedef __attribute__((ext_vector_type(8)))  float    v8f;

// ---------------------------------------------------------------- WMMA core
__device__ __forceinline__ v8f wmma16(v16h a, v16h b, v8f c) {
  // v_wmma_f32_16x16x32_f16: D = A(16x32,f16) x B(32x16,f16) + C(16x16,f32)
  return __builtin_amdgcn_wmma_f32_16x16x32_f16(false, a, false, b, (short)0, c,
                                                false, false);
}

// A fragment (16x32 f16): lanes 0-15 hold M=lane, K 0-15;
// lanes 16-31 same M with K 16-31 (contiguous, matches B fragments).
__device__ __forceinline__ v16h frag_a(const _Float16* __restrict__ base, int ld,
                                       int lane) {
  const int m = lane & 15;
  const int hf = lane >> 4;
  const _Float16* row = base + m * ld;
  v16h a;
#pragma unroll
  for (int v = 0; v < 8; ++v) {
    int k = hf * 16 + v * 2;
    a[2 * v] = row[k];
    a[2 * v + 1] = row[k + 1];
  }
  return a;
}

// B fragment (32x16 f16) from N-major storage: B[k][n] = base[n*ld + k]
__device__ __forceinline__ v16h frag_bT(const _Float16* __restrict__ base, int ld,
                                        int lane) {
  const int n = lane & 15;
  const int koff = (lane >> 4) * 16;
  const _Float16* row = base + n * ld + koff;
  v16h b;
#pragma unroll
  for (int i = 0; i < 16; ++i) b[i] = row[i];
  return b;
}

// B fragment (32x16 f16) from K-major storage: B[k][n] = base[k*ld + n]
__device__ __forceinline__ v16h frag_bN(const _Float16* __restrict__ base, int ld,
                                        int lane) {
  const int n = lane & 15;
  const int koff = (lane >> 4) * 16;
  v16h b;
#pragma unroll
  for (int i = 0; i < 16; ++i) b[i] = base[(koff + i) * ld + n];
  return b;
}

// ------------------------------------------------------------ generic GEMM
// C[M,N] = op( A[M,K](f32) @ W[N,K]^T(f16) + bias[N] (+ R[M,N]) )
// flags: bit0 = relu, bit1 = residual add
// Double-buffered LDS; W tiles fetched with CDNA5 async global->LDS DMA
// (ASYNCcnt), A tiles staged via registers with on-the-fly f32->f16 convert.
#define BM 128
#define BN 128
#define BKt 32
#define ATILE (BM * BKt)  // 4096 halfs = 8 KB
#define WTILE (BN * BKt)  // 4096 halfs = 8 KB

__global__ __launch_bounds__(256) void k_gemm(
    const float* __restrict__ A, const _Float16* __restrict__ W,
    const float* __restrict__ bias, const float* __restrict__ R,
    float* __restrict__ C, int M, int N, int K, int flags) {
  // single __shared__ symbol -> group-segment offset 0; layout: A0 A1 W0 W1
  // (no pointer arrays into LDS: lld rejects addrspacecast initializers)
  __shared__ _Float16 smem[2 * ATILE + 2 * WTILE];
  const unsigned WLDS0 = 2u * ATILE * 2u;  // byte offset of W0 in LDS

  const int t = threadIdx.x;
  const int lane = t & 31;
  const int w = t >> 5;   // 8 waves
  const int wm = w >> 2;  // 0..1 : 64-row stripe
  const int wn = w & 3;   // 0..3 : 32-col stripe
  const int m0 = blockIdx.y * BM;
  const int n0 = blockIdx.x * BN;

  const int ar = t >> 3, ac4 = (t & 7) * 4;  // A staging coords (4 x float4)
  const int wr = t >> 2, wc8 = (t & 3) * 8;  // W staging coords (2 x 16B)

  const v8f vzero = {0, 0, 0, 0, 0, 0, 0, 0};
  v8f acc[4][2];
#pragma unroll
  for (int i = 0; i < 4; ++i)
#pragma unroll
    for (int j = 0; j < 2; ++j) acc[i][j] = vzero;

  float4 areg[4];

  auto issue_w_async = [&](int k0, int buf) {
#pragma unroll
    for (int p = 0; p < 2; ++p) {
      int r = p * 64 + wr;
      unsigned lds = WLDS0 + (unsigned)buf * (WTILE * 2u) +
                     ((unsigned)(r * BKt + wc8)) * 2u;
      unsigned long long ga =
          (unsigned long long)(size_t)(W + (size_t)(n0 + r) * K + k0 + wc8);
      // CDNA5 async DMA: global -> LDS, 16B per lane, tracked by ASYNCcnt
      asm volatile("global_load_async_to_lds_b128 %0, %1, off"
                   :: "v"(lds), "v"(ga) : "memory");
    }
  };
  auto load_a = [&](int k0) {
#pragma unroll
    for (int p = 0; p < 4; ++p)
      areg[p] = *(const float4*)(A + (size_t)(m0 + p * 32 + ar) * K + k0 + ac4);
  };
  auto store_a = [&](int buf) {
    _Float16* ab = smem + buf * ATILE;
#pragma unroll
    for (int p = 0; p < 4; ++p) {
      _Float16* d = ab + (p * 32 + ar) * BKt + ac4;
      d[0] = (_Float16)areg[p].x; d[1] = (_Float16)areg[p].y;
      d[2] = (_Float16)areg[p].z; d[3] = (_Float16)areg[p].w;
    }
  };

  // prologue: tile 0 into buffer 0
  issue_w_async(0, 0);
  load_a(0);
  store_a(0);
  asm volatile("s_wait_asynccnt 0x0" ::: "memory");
  __syncthreads();

  const int nk = K / BKt;
  for (int kt = 0; kt < nk; ++kt) {
    const int buf = kt & 1;
    if (kt + 1 < nk) {
      issue_w_async((kt + 1) * BKt, buf ^ 1);  // overlap DMA with WMMA
      load_a((kt + 1) * BKt);                  // overlap loads with WMMA
      __builtin_prefetch(A + (size_t)(m0 + ar) * K + (kt + 2) * BKt, 0, 0);
    }

    const _Float16* ab = smem + buf * ATILE;
    const _Float16* wb = smem + 2 * ATILE + buf * WTILE;
    v16h af[4];
#pragma unroll
    for (int mt = 0; mt < 4; ++mt)
      af[mt] = frag_a(ab + (wm * 64 + mt * 16) * BKt, BKt, lane);
    v16h bf[2];
#pragma unroll
    for (int nt = 0; nt < 2; ++nt)
      bf[nt] = frag_bT(wb + (wn * 32 + nt * 16) * BKt, BKt, lane);
#pragma unroll
    for (int mt = 0; mt < 4; ++mt)
#pragma unroll
      for (int nt = 0; nt < 2; ++nt)
        acc[mt][nt] = wmma16(af[mt], bf[nt], acc[mt][nt]);

    if (kt + 1 < nk) {
      store_a(buf ^ 1);  // other buffer: no reader this step
      asm volatile("s_wait_asynccnt 0x0" ::: "memory");
      __syncthreads();   // single barrier per K-step
    }
  }

  const int rh = lane >> 4;
  const int nlane = lane & 15;
#pragma unroll
  for (int mt = 0; mt < 4; ++mt) {
#pragma unroll
    for (int nt = 0; nt < 2; ++nt) {
      const int col = n0 + wn * 32 + nt * 16 + nlane;
      const float bv = bias ? bias[col] : 0.0f;
#pragma unroll
      for (int r = 0; r < 8; ++r) {
        const int row = m0 + wm * 64 + mt * 16 + rh * 8 + r;
        const size_t idx = (size_t)row * N + col;
        float v = acc[mt][nt][r] + bv;
        if (flags & 2) v += R[idx];
        if (flags & 1) v = fmaxf(v, 0.0f);
        C[idx] = v;
      }
    }
  }
}

// ----------------------------------------------- fused encoder self-attention
// One block per (batch, head): S=128, DK=32. Q.K^T -> softmax -> P.V, all WMMA.
__global__ __launch_bounds__(256) void k_attn(const float* __restrict__ qkv,
                                              float* __restrict__ ctx) {
  __shared__ _Float16 QVs[Ssz * DKsz];  // Q, later reused for V
  __shared__ _Float16 Ks[Ssz * DKsz];
  __shared__ _Float16 Ps[Ssz * Ssz];    // scores / probabilities (f16)
  const int b = blockIdx.x / Hh;
  const int hd = blockIdx.x % Hh;
  const int t = threadIdx.x;
  const int lane = t & 31;
  const int w = t >> 5;
  const size_t qbase = (size_t)(b * Ssz) * (3 * Esz) + hd * DKsz;
  const int sr = t >> 3, sc4 = (t & 7) * 4;

  {  // stage Q,K: issue all loads first (MLP), then convert+store
    float4 vq[4], vk[4];
#pragma unroll
    for (int p = 0; p < 4; ++p) {
      int s = p * 32 + sr;
      vq[p] = *(const float4*)(qkv + qbase + (size_t)s * 768 + sc4);
      vk[p] = *(const float4*)(qkv + qbase + Esz + (size_t)s * 768 + sc4);
    }
#pragma unroll
    for (int p = 0; p < 4; ++p) {
      int s = p * 32 + sr;
      _Float16* dq = QVs + s * DKsz + sc4;
      _Float16* dk = Ks + s * DKsz + sc4;
      dq[0] = (_Float16)vq[p].x; dq[1] = (_Float16)vq[p].y;
      dq[2] = (_Float16)vq[p].z; dq[3] = (_Float16)vq[p].w;
      dk[0] = (_Float16)vk[p].x; dk[1] = (_Float16)vk[p].y;
      dk[2] = (_Float16)vk[p].z; dk[3] = (_Float16)vk[p].w;
    }
  }
  __syncthreads();

  {  // scores: wave w -> rows [16w,16w+16), 8 column tiles, K=32 (1 WMMA each)
    const v8f vzero = {0, 0, 0, 0, 0, 0, 0, 0};
    const v16h aq = frag_a(QVs + (w * 16) * DKsz, DKsz, lane);
    const float scale = 0.17677669529663687f;  // 1/sqrt(32)
    const int rh = lane >> 4, nlane = lane & 15;
#pragma unroll
    for (int nt = 0; nt < 8; ++nt) {
      v16h bk = frag_bT(Ks + (nt * 16) * DKsz, DKsz, lane);
      v8f sc = wmma16(aq, bk, vzero);
#pragma unroll
      for (int r = 0; r < 8; ++r)
        Ps[(w * 16 + rh * 8 + r) * Ssz + nt * 16 + nlane] =
            (_Float16)(sc[r] * scale);
    }
  }
  __syncthreads();

  for (int rr = 0; rr < 16; ++rr) {  // softmax, wave per 16-row stripe
    const int row = w * 16 + rr;
    float x[4];
#pragma unroll
    for (int i = 0; i < 4; ++i) x[i] = (float)Ps[row * Ssz + lane + 32 * i];
    float mx = fmaxf(fmaxf(x[0], x[1]), fmaxf(x[2], x[3]));
#pragma unroll
    for (int off = 16; off > 0; off >>= 1) mx = fmaxf(mx, __shfl_xor(mx, off, 32));
    float sum = 0.0f;
#pragma unroll
    for (int i = 0; i < 4; ++i) { x[i] = __expf(x[i] - mx); sum += x[i]; }
#pragma unroll
    for (int off = 16; off > 0; off >>= 1) sum += __shfl_xor(sum, off, 32);
    const float inv = 1.0f / sum;
#pragma unroll
    for (int i = 0; i < 4; ++i)
      Ps[row * Ssz + lane + 32 * i] = (_Float16)(x[i] * inv);
  }
  __syncthreads();

  {  // stage V into QVs (loads first, then convert+store)
    float4 vv[4];
#pragma unroll
    for (int p = 0; p < 4; ++p)
      vv[p] = *(const float4*)(qkv + qbase + 2 * Esz +
                               (size_t)(p * 32 + sr) * 768 + sc4);
#pragma unroll
    for (int p = 0; p < 4; ++p) {
      _Float16* dv = QVs + (p * 32 + sr) * DKsz + sc4;
      dv[0] = (_Float16)vv[p].x; dv[1] = (_Float16)vv[p].y;
      dv[2] = (_Float16)vv[p].z; dv[3] = (_Float16)vv[p].w;
    }
  }
  __syncthreads();

  const v8f vzero = {0, 0, 0, 0, 0, 0, 0, 0};
  v8f oc[2] = {vzero, vzero};
  for (int kk = 0; kk < 8; ++kk) {  // ctx = P @ V, 8 k-steps of 32 (K=256)
    v16h ap = frag_a(Ps + (w * 16) * Ssz + kk * 32, Ssz, lane);
#pragma unroll
    for (int nt = 0; nt < 2; ++nt) {
      v16h bv = frag_bN(QVs + (kk * 32) * DKsz + nt * 16, DKsz, lane);
      oc[nt] = wmma16(ap, bv, oc[nt]);
    }
  }
  const int rh = lane >> 4, nlane = lane & 15;
#pragma unroll
  for (int nt = 0; nt < 2; ++nt)
#pragma unroll
    for (int r = 0; r < 8; ++r)
      ctx[((size_t)(b * Ssz + w * 16 + rh * 8 + r)) * Esz + hd * DKsz +
          nt * 16 + nlane] = oc[nt][r];
}

// ------------------------------------------------------------ small kernels
__global__ void k_cvt(const float* __restrict__ s, _Float16* __restrict__ d,
                      int n) {
  int i = blockIdx.x * 256 + threadIdx.x;
  if (i < n) d[i] = (_Float16)s[i];
}

__global__ void k_embed(const float* __restrict__ in, const float* __restrict__ Wx,
                        const float* __restrict__ bx, float* __restrict__ h) {
  int bs = blockIdx.x, e = threadIdx.x;
  float x0 = in[bs * 2], x1 = in[bs * 2 + 1];
  h[(size_t)bs * Esz + e] = x0 * Wx[e * 2] + x1 * Wx[e * 2 + 1] + bx[e];
}

__global__ void k_bn_stats(const float* __restrict__ X, float* __restrict__ mu,
                           float* __restrict__ rstd) {
  __shared__ float s1[256], s2[256];
  int c = blockIdx.x, t = threadIdx.x;
  float a = 0.0f, b = 0.0f;
  for (int r = t; r < Bsz * Ssz; r += 256) {
    float x = X[(size_t)r * Esz + c];
    a += x; b += x * x;
  }
  s1[t] = a; s2[t] = b; __syncthreads();
  for (int off = 128; off > 0; off >>= 1) {
    if (t < off) { s1[t] += s1[t + off]; s2[t] += s2[t + off]; }
    __syncthreads();
  }
  if (t == 0) {
    float m = s1[0] / (float)(Bsz * Ssz);
    float v = s2[0] / (float)(Bsz * Ssz) - m * m;
    mu[c] = m;
    rstd[c] = rsqrtf(v + 1e-5f);
  }
}

__global__ void k_bn_apply(float* __restrict__ X, const float* __restrict__ mu,
                           const float* __restrict__ rstd,
                           const float* __restrict__ g,
                           const float* __restrict__ bb) {
  int c = threadIdx.x;
  size_t i = (size_t)blockIdx.x * Esz + c;
  X[i] = (X[i] - mu[c]) * rstd[c] * g[c] + bb[c];
}

__global__ void k_rowsum(const float* __restrict__ X, float* __restrict__ out) {
  int b = blockIdx.x, e = threadIdx.x;
  float a = 0.0f;
  for (int s = 0; s < Ssz; ++s) a += X[((size_t)(b * Ssz + s)) * Esz + e];
  out[(size_t)b * Esz + e] = a;
}

__global__ void k_vw0(const float* __restrict__ Wv, const float* __restrict__ bv,
                      const float* __restrict__ iw, float* __restrict__ vwch) {
  __shared__ float s[Esz];
  int b = blockIdx.x, e = threadIdx.x;
  s[e] = iw[e]; __syncthreads();
  float a = bv[e];
  for (int k = 0; k < Esz; ++k) a += Wv[(size_t)e * Esz + k] * s[k];
  vwch[(size_t)b * Esz + e] = a;
}

__global__ void k_tanh2(const float* __restrict__ cv, const float* __restrict__ lv,
                        float* __restrict__ ta, float* __restrict__ tb) {
  size_t i = (size_t)blockIdx.x * 256 + threadIdx.x;
  ta[i] = tanhf(cv[i]);
  tb[i] = tanhf(lv[i]);
}

__global__ void k_add(float* __restrict__ u, const float* __restrict__ v) {
  size_t i = (size_t)blockIdx.x * 256 + threadIdx.x;
  u[i] += v[i];
}

// masked multi-head glimpse attention, one block per batch element
__global__ __launch_bounds__(256) void k_glimpse(
    const float* __restrict__ gq, const float* __restrict__ gk,
    const float* __restrict__ gv, const int* __restrict__ mask,
    float* __restrict__ out) {
  __shared__ float qs[Esz];
  __shared__ float gl[Hh * Ssz];
  int b = blockIdx.x, t = threadIdx.x, lane = t & 31, w = t >> 5;
  qs[t] = gq[(size_t)b * Esz + t];
  __syncthreads();
  const float scale = 0.17677669529663687f;
#pragma unroll
  for (int p = 0; p < 4; ++p) {
    int idx = p * 256 + t;
    int hh = idx >> 7, s = idx & 127;
    const float* kr = gk + ((size_t)(b * Ssz + s)) * Esz + hh * DKsz;
    float a = 0.0f;
#pragma unroll
    for (int d = 0; d < DKsz; ++d) a += qs[hh * DKsz + d] * kr[d];
    a *= scale;
    if (mask[b * Ssz + s]) a = NEGV;
    gl[hh * Ssz + s] = a;
  }
  __syncthreads();
  {  // softmax: wave w owns head w (8 waves, 8 heads)
    float x[4];
#pragma unroll
    for (int i = 0; i < 4; ++i) x[i] = gl[w * Ssz + lane + 32 * i];
    float mx = fmaxf(fmaxf(x[0], x[1]), fmaxf(x[2], x[3]));
#pragma unroll
    for (int off = 16; off > 0; off >>= 1) mx = fmaxf(mx, __shfl_xor(mx, off, 32));
    float sum = 0.0f;
#pragma unroll
    for (int i = 0; i < 4; ++i) { x[i] = __expf(x[i] - mx); sum += x[i]; }
#pragma unroll
    for (int off = 16; off > 0; off >>= 1) sum += __shfl_xor(sum, off, 32);
    float inv = 1.0f / sum;
#pragma unroll
    for (int i = 0; i < 4; ++i) gl[w * Ssz + lane + 32 * i] = x[i] * inv;
  }
  __syncthreads();
  int hh = t >> 5, d = t & 31;
  float a = 0.0f;
  for (int s = 0; s < Ssz; ++s)
    a += gl[hh * Ssz + s] * gv[((size_t)(b * Ssz + s)) * Esz + hh * DKsz + d];
  out[(size_t)b * Esz + t] = a;
}

// pointer attention + argmax + state update, one block (128 thr) per batch elem
__global__ __launch_bounds__(128) void k_pointer(
    const float* __restrict__ pq, const float* __restrict__ pkb,
    int* __restrict__ mask, const float* __restrict__ hcho,
    const float* __restrict__ hmem, const float* __restrict__ hvw,
    float* __restrict__ cv, float* __restrict__ lv, float* __restrict__ vwch,
    float* __restrict__ out, int step) {
  __shared__ float qs[Esz];
  __shared__ float lbuf[Ssz];
  __shared__ float red[Ssz];
  __shared__ int redi[Ssz];
  int b = blockIdx.x, t = threadIdx.x;
  qs[t] = pq[(size_t)b * Esz + t];
  qs[t + 128] = pq[(size_t)b * Esz + t + 128];
  __syncthreads();
  const float* kr = pkb + ((size_t)(b * Ssz + t)) * Esz;
  float dot = 0.0f;
  for (int e = 0; e < Esz; ++e) dot += qs[e] * kr[e];
  float l = 10.0f * tanhf(dot * 0.0625f);  // CLIP * tanh(dot / sqrt(E))
  if (mask[b * Ssz + t]) l = NEGV;
  lbuf[t] = l; red[t] = l; redi[t] = t;
  __syncthreads();
  for (int off = 64; off > 0; off >>= 1) {  // argmax (first index on ties)
    if (t < off) {
      float o = red[t + off]; int oi = redi[t + off];
      if (o > red[t] || (o == red[t] && oi < redi[t])) { red[t] = o; redi[t] = oi; }
    }
    __syncthreads();
  }
  const int chosen = redi[0];
  const float mx = red[0];
  __syncthreads();
  red[t] = __expf(l - mx);
  __syncthreads();
  for (int off = 64; off > 0; off >>= 1) {
    if (t < off) red[t] += red[t + off];
    __syncthreads();
  }
  if (t == 0) {
    float lse = mx + __logf(red[0]);
    out[(size_t)b * Ssz + step] = lbuf[chosen] - lse;                 // logp
    out[(size_t)Bsz * Ssz + (size_t)b * Ssz + step] = (float)chosen;  // index
    mask[b * Ssz + chosen] = 1;
  }
  const size_t base = ((size_t)(b * Ssz + chosen)) * Esz;
  for (int e = t; e < Esz; e += 128) {
    cv[(size_t)b * Esz + e] += hcho[base + e];
    lv[(size_t)b * Esz + e] -= hmem[base + e];
    vwch[(size_t)b * Esz + e] = hvw[base + e];
  }
}

// ------------------------------------------------------------------- driver
extern "C" void kernel_launch(void* const* d_in, const int* in_sizes, int n_in,
                              void* d_out, int out_size, void* d_ws,
                              size_t ws_size, hipStream_t stream) {
  (void)in_sizes; (void)n_in; (void)out_size; (void)ws_size;
  // JAX pytree flatten order (dict keys sorted; each linear = {b, w})
  const float* x_in   = (const float*)d_in[0];
  const float* cho_b  = (const float*)d_in[1];
  const float* cho_w  = (const float*)d_in[2];
  const float* emb_b  = (const float*)d_in[3];
  const float* emb_w  = (const float*)d_in[4];
  const float* gkk_b  = (const float*)d_in[5];
  const float* gkk_w  = (const float*)d_in[6];
  const float* go_b   = (const float*)d_in[7];
  const float* go_w   = (const float*)d_in[8];
  const float* gqq_b  = (const float*)d_in[9];
  const float* gqq_w  = (const float*)d_in[10];
  const float* gvv_b  = (const float*)d_in[11];
  const float* gvv_w  = (const float*)d_in[12];
  const float* h1_b   = (const float*)d_in[13];
  const float* h1_w   = (const float*)d_in[14];
  const float* h2_b   = (const float*)d_in[15];
  const float* h2_w   = (const float*)d_in[16];
  // d_in[17],[18]: h_context (computed-but-unused in reference) -> skipped
  const float* hq_b   = (const float*)d_in[19];
  const float* hq_w   = (const float*)d_in[20];
  const float* init_w = (const float*)d_in[21];
  struct Layer {
    const float *bn_b, *bn_g, *ff1_b, *ff1_w, *ff2_b, *ff2_w, *ip_b, *ip_w,
        *op_b, *op_w;
  };
  Layer L[2];
  for (int i = 0; i < 2; ++i) {
    int o = 22 + i * 10;
    L[i].bn_b = (const float*)d_in[o + 0]; L[i].bn_g = (const float*)d_in[o + 1];
    L[i].ff1_b = (const float*)d_in[o + 2]; L[i].ff1_w = (const float*)d_in[o + 3];
    L[i].ff2_b = (const float*)d_in[o + 4]; L[i].ff2_w = (const float*)d_in[o + 5];
    L[i].ip_b = (const float*)d_in[o + 6]; L[i].ip_w = (const float*)d_in[o + 7];
    L[i].op_b = (const float*)d_in[o + 8]; L[i].op_w = (const float*)d_in[o + 9];
  }
  const float* mem_b = (const float*)d_in[42];
  const float* mem_w = (const float*)d_in[43];
  const float* pkk_b = (const float*)d_in[44];
  const float* pkk_w = (const float*)d_in[45];
  const float* pqq_b = (const float*)d_in[46];
  const float* pqq_w = (const float*)d_in[47];
  const float* vw_b  = (const float*)d_in[48];
  const float* vw_w  = (const float*)d_in[49];

  // ---- workspace carve (256B aligned). Peak ~280 MB.
  char* ws = (char*)d_ws;
  size_t off = 0;
  auto alloc = [&](size_t bytes) -> void* {
    void* p = ws + off;
    off += (bytes + 255) & ~(size_t)255;
    return p;
  };
  const size_t BSE = (size_t)Bsz * Ssz * Esz;  // 8.39M elements
  const size_t BE = (size_t)Bsz * Esz;
  float* h    = (float*)alloc(BSE * 4);
  float* qkv  = (float*)alloc(BSE * 3 * 4);
  float* ctxb = (float*)alloc(BSE * 4);
  float* x1   = (float*)alloc(BSE * 4);
  float* ffh  = (float*)alloc((size_t)Bsz * Ssz * HIDsz * 4);
  // decode-phase aliases (encoder buffers dead by then)
  float* gkb = qkv;            // glimpse K proj   [B,S,E]
  float* gvb = qkv + BSE;      // glimpse V proj   [B,S,E]
  float* pkb = qkv + 2 * BSE;  // pointer K proj   [B,S,E]
  float* hmem = ffh;           // memory(h)        [B,S,E]
  float* hcho = ffh + BSE;     // chosen(h)        [B,S,E]
  float* hvw = x1;             // v_weight(h)      [B,S,E]
  float* mu = (float*)alloc(Esz * 4);
  float* rstd = (float*)alloc(Esz * 4);
  float* cv = (float*)alloc(BE * 4);
  float* lv = (float*)alloc(BE * 4);
  float* qv = (float*)alloc(BE * 4);
  float* gqb = (float*)alloc(BE * 4);
  float* gct = (float*)alloc(BE * 4);
  float* nqb = (float*)alloc(BE * 4);
  float* pqb = (float*)alloc(BE * 4);
  float* ub = (float*)alloc(BE * 4);
  float* ta = (float*)alloc(BE * 4);
  float* tb = (float*)alloc(BE * 4);
  float* vwch = (float*)alloc(BE * 4);
  int* mask = (int*)alloc((size_t)Bsz * Ssz * 4);

  // ---- one-time f16 weight conversion (feeds the WMMA B operand)
  auto cvt = [&](const float* src, int n) -> _Float16* {
    _Float16* d = (_Float16*)alloc((size_t)n * 2);
    k_cvt<<<(n + 255) / 256, 256, 0, stream>>>(src, d, n);
    return d;
  };
  _Float16 *ip16[2], *op16[2], *f116[2], *f216[2];
  for (int i = 0; i < 2; ++i) {
    ip16[i] = cvt(L[i].ip_w, 768 * 256);
    op16[i] = cvt(L[i].op_w, 256 * 256);
    f116[i] = cvt(L[i].ff1_w, 512 * 256);
    f216[i] = cvt(L[i].ff2_w, 256 * 512);
  }
  _Float16* gk16 = cvt(gkk_w, 65536);
  _Float16* gv16 = cvt(gvv_w, 65536);
  _Float16* gq16 = cvt(gqq_w, 65536);
  _Float16* go16 = cvt(go_w, 65536);
  _Float16* pk16 = cvt(pkk_w, 65536);
  _Float16* pq16 = cvt(pqq_w, 65536);
  _Float16* mem16 = cvt(mem_w, 65536);
  _Float16* cho16 = cvt(cho_w, 65536);
  _Float16* vw16 = cvt(vw_w, 65536);
  _Float16* h116 = cvt(h1_w, 65536);
  _Float16* h216 = cvt(h2_w, 65536);
  _Float16* hq16 = cvt(hq_w, 65536);

  auto gemm = [&](const float* A, const _Float16* Wg, const float* bias,
                  const float* R, float* Cc, int M, int N, int K, int flags) {
    dim3 g(N / 128, M / 128);
    k_gemm<<<g, 256, 0, stream>>>(A, Wg, bias, R, Cc, M, N, K, flags);
  };

  (void)hipMemsetAsync(mask, 0, (size_t)Bsz * Ssz * 4, stream);
  (void)hipMemsetAsync(cv, 0, BE * 4, stream);

  // ---- encoder
  const int Mfull = Bsz * Ssz;
  k_embed<<<Mfull, Esz, 0, stream>>>(x_in, emb_w, emb_b, h);
  for (int i = 0; i < 2; ++i) {
    gemm(h, ip16[i], L[i].ip_b, nullptr, qkv, Mfull, 3 * Esz, Esz, 0);
    k_attn<<<Bsz * Hh, 256, 0, stream>>>(qkv, ctxb);
    gemm(ctxb, op16[i], L[i].op_b, h, x1, Mfull, Esz, Esz, 2);          // +res
    gemm(x1, f116[i], L[i].ff1_b, nullptr, ffh, Mfull, HIDsz, Esz, 1);  // relu
    gemm(ffh, f216[i], L[i].ff2_b, x1, h, Mfull, Esz, HIDsz, 2);        // +res
    k_bn_stats<<<Esz, 256, 0, stream>>>(h, mu, rstd);
    k_bn_apply<<<Mfull, Esz, 0, stream>>>(h, mu, rstd, L[i].bn_g, L[i].bn_b);
  }

  // ---- decode precompute: step-invariant projections hoisted out of the loop
  gemm(h, gk16, gkk_b, nullptr, gkb, Mfull, Esz, Esz, 0);
  gemm(h, gv16, gvv_b, nullptr, gvb, Mfull, Esz, Esz, 0);
  gemm(h, pk16, pkk_b, nullptr, pkb, Mfull, Esz, Esz, 0);
  gemm(h, mem16, mem_b, nullptr, hmem, Mfull, Esz, Esz, 0);
  gemm(h, cho16, cho_b, nullptr, hcho, Mfull, Esz, Esz, 0);
  gemm(h, vw16, vw_b, nullptr, hvw, Mfull, Esz, Esz, 0);
  k_rowsum<<<Bsz, Esz, 0, stream>>>(hmem, lv);               // lv0
  k_vw0<<<Bsz, Esz, 0, stream>>>(vw_w, vw_b, init_w, vwch);  // vw0 broadcast

  auto update_q = [&]() {
    k_tanh2<<<Bsz, Esz, 0, stream>>>(cv, lv, ta, tb);
    gemm(ta, h116, h1_b, nullptr, ub, Bsz, Esz, Esz, 0);
    gemm(tb, h216, h2_b, ub, ub, Bsz, Esz, Esz, 2);  // += previous (residual)
    k_add<<<Bsz, Esz, 0, stream>>>(ub, vwch);
    gemm(ub, hq16, hq_b, nullptr, qv, Bsz, Esz, Esz, 0);
  };
  update_q();  // q0 (tanh(cv0)=0 handled naturally)

  // ---- 128 sequential decode steps
  float* out = (float*)d_out;
  for (int t = 0; t < Ssz; ++t) {
    gemm(qv, gq16, gqq_b, nullptr, gqb, Bsz, Esz, Esz, 0);
    k_glimpse<<<Bsz, 256, 0, stream>>>(gqb, gkb, gvb, mask, gct);
    gemm(gct, go16, go_b, nullptr, nqb, Bsz, Esz, Esz, 0);
    gemm(nqb, pq16, pqq_b, nullptr, pqb, Bsz, Esz, Esz, 0);
    k_pointer<<<Bsz, 128, 0, stream>>>(pqb, pkb, mask, hcho, hmem, hvw, cv, lv,
                                       vwch, out, t);
    update_q();
  }
}
